// DCNet_55954833932882
// MI455X (gfx1250) — compile-verified, matrix operand order checked by reading, not run on hardware
//
#include <hip/hip_runtime.h>
#include <hip/hip_bf16.h>

#define NB 16384
#define HD 512
#define BM 64
#define BN 128
#define BK 32
#define ASTR 40    // padded k-stride for A tile in LDS (gemm kernels)
#define BSTR 40    // padded k-stride for B^T tile in LDS
#define AST2 520   // padded k-stride for full-K A panel in LDS (dfc1)

typedef __attribute__((ext_vector_type(16))) __bf16 v16bf;
typedef __attribute__((ext_vector_type(8)))  float  v8f;
typedef __attribute__((ext_vector_type(4)))  int    v4i;

__device__ inline __bf16 f2bf(float f) {
    unsigned u = __builtin_bit_cast(unsigned, f);
    unsigned r = (u + 0x7FFFu + ((u >> 16) & 1u)) >> 16;
    unsigned short s = (unsigned short)r;
    return __builtin_bit_cast(__bf16, s);
}

struct bf16x16_bits { uint4 lo, hi; };
__device__ inline v16bf ld_frag16(const __bf16* p) {
    bf16x16_bits t;
    t.lo = *reinterpret_cast<const uint4*>(p);
    t.hi = *reinterpret_cast<const uint4*>(p + 8);
    return __builtin_bit_cast(v16bf, t);
}

// ---- 16B global->LDS copy: async (ASYNCcnt) when the gfx1250 builtin exists ----
#if __has_builtin(__builtin_amdgcn_global_load_async_to_lds_b128) && \
    __has_builtin(__builtin_amdgcn_s_wait_asynccnt)
#define HAS_ASYNC_LDS 1
#else
#define HAS_ASYNC_LDS 0
#endif

__device__ inline void cp16(const __bf16* __restrict__ g, __bf16* l) {
#if HAS_ASYNC_LDS
    __builtin_amdgcn_global_load_async_to_lds_b128(
        (__attribute__((address_space(1))) v4i*)g,
        (__attribute__((address_space(3))) v4i*)l, 0, 0);
#else
    *reinterpret_cast<uint4*>(l) = *reinterpret_cast<const uint4*>(g);
#endif
}
__device__ inline void cp_wait() {
#if HAS_ASYNC_LDS
    __builtin_amdgcn_s_wait_asynccnt(0);
#endif
}

// ---------------- treatment basis: tb[b, 25] ----------------
__global__ void basis_kernel(const float* __restrict__ t, float* __restrict__ tb, int Bn) {
    int b = blockIdx.x * blockDim.x + threadIdx.x;
    if (b >= Bn) return;
    float t0 = t[b * 2 + 0], t1 = t[b * 2 + 1];
    float b0[5], b1[5], r;
    b0[0] = 1.f; b0[1] = t0; b0[2] = t0 * t0;
    r = fmaxf(t0 - 0.33f, 0.f); b0[3] = r * r;
    r = fmaxf(t0 - 0.66f, 0.f); b0[4] = r * r;
    b1[0] = 1.f; b1[1] = t1; b1[2] = t1 * t1;
    r = fmaxf(t1 - 0.33f, 0.f); b1[3] = r * r;
    r = fmaxf(t1 - 0.66f, 0.f); b1[4] = r * r;
    #pragma unroll
    for (int i = 0; i < 5; ++i)
        #pragma unroll
        for (int j = 0; j < 5; ++j)
            tb[(size_t)b * 25 + i * 5 + j] = b0[i] * b1[j];
}

// ---------------- fp32 -> bf16 converters ----------------
__global__ void cvt_kernel(const float* __restrict__ s, __bf16* __restrict__ d, size_t n) {
    size_t i = (size_t)blockIdx.x * blockDim.x + threadIdx.x;
    if (i < n) d[i] = f2bf(s[i]);
}

// src [K,N] row-major f32  ->  dst [N,K] row-major bf16 (i.e. B^T)
__global__ void cvt_transpose(const float* __restrict__ s, __bf16* __restrict__ d,
                              int K, int N) {
    int i = blockIdx.x * blockDim.x + threadIdx.x;
    if (i >= N * K) return;
    int n = i / K, k = i - n * K;
    d[i] = f2bf(s[(size_t)k * N + n]);
}

// dfc1_w [i,o,d] -> wt[d][o][i] bf16  (per-d B^T planes, row o contiguous in i)
__global__ void cvt_dfc1w(const float* __restrict__ w, __bf16* __restrict__ wt) {
    int t = blockIdx.x * blockDim.x + threadIdx.x;   // t = o*512 + i
    int o = t >> 9, i = t & 511;
    const float* src = w + ((size_t)i * HD + o) * 25;   // 25 contiguous d values
    #pragma unroll
    for (int d = 0; d < 25; ++d)
        wt[(size_t)d * HD * HD + t] = f2bf(src[d]);
}

// ------------- generic bf16 WMMA GEMM: C = act(A@B + bias), B given as B^T -------------
__global__ __launch_bounds__(256) void gemm_bf16(
    const __bf16* __restrict__ A,   // [M,K] row-major
    const __bf16* __restrict__ BT,  // [N,K] row-major (transposed B)
    const float* __restrict__ bias, float* __restrict__ outF,
    __bf16* __restrict__ outB, int M, int N, int K, int relu)
{
    __shared__ __bf16 Alds[BM * ASTR];
    __shared__ __bf16 Blds[BN * BSTR];

    const int tid  = threadIdx.x;
    const int lane = tid & 31;
    const int wave = tid >> 5;
    const int wm = (wave & 1) * 32;
    const int wn = (wave >> 1) * 32;
    const int bm = blockIdx.x * BM;
    const int bn = blockIdx.y * BN;
    const int lrow  = lane & 15;
    const int khalf = (lane >> 4) * 16;

    v8f acc[2][2] = {};

    for (int k0 = 0; k0 < K; k0 += BK) {
        {   // stage A: 64x32, one 16B chunk per thread
            int r = tid >> 2;
            int c = (tid & 3) * 8;
            cp16(A + (size_t)(bm + r) * K + k0 + c, &Alds[r * ASTR + c]);
        }
        {   // stage B^T: 128 cols x 32 k, two 16B chunks per thread
            int n = tid >> 1;
            int kseg = (tid & 1) * 16;
            const __bf16* src = BT + (size_t)(bn + n) * K + k0 + kseg;
            cp16(src,     &Blds[n * BSTR + kseg]);
            cp16(src + 8, &Blds[n * BSTR + kseg + 8]);
        }
        cp_wait();
        __syncthreads();
        #pragma unroll
        for (int mi = 0; mi < 2; ++mi) {
            v16bf a = ld_frag16(&Alds[(wm + mi * 16 + lrow) * ASTR + khalf]);
            #pragma unroll
            for (int ni = 0; ni < 2; ++ni) {
                v16bf b = ld_frag16(&Blds[(wn + ni * 16 + lrow) * BSTR + khalf]);
                acc[mi][ni] = __builtin_amdgcn_wmma_f32_16x16x32_bf16(
                    false, a, false, b, (short)0, acc[mi][ni], false, false);
            }
        }
        __syncthreads();
    }

    #pragma unroll
    for (int mi = 0; mi < 2; ++mi)
        #pragma unroll
        for (int ni = 0; ni < 2; ++ni) {
            int col = bn + wn + ni * 16 + lrow;
            float bs = bias ? bias[col] : 0.f;
            #pragma unroll
            for (int v = 0; v < 8; ++v) {
                int row = bm + wm + mi * 16 + v + ((lane >> 4) << 3);
                float r0 = acc[mi][ni][v] + bs;
                if (relu) r0 = fmaxf(r0, 0.f);
                if (outF) outF[(size_t)row * N + col] = r0;
                if (outB) outB[(size_t)row * N + col] = f2bf(r0);
            }
        }
}

// ------- dfc1: h1 = relu( sum_d tb[b,d] * (hidden @ w_d + b1[:,d]) ) -------
// A panel (64 x 512) staged in LDS ONCE; per-d B^T tiles streamed.
__global__ __launch_bounds__(256) void dfc1_gemm(
    const __bf16* __restrict__ A,      // hidden bf16 [M, 512]
    const __bf16* __restrict__ Wt,     // [25][512][512] bf16, per-d B^T
    const float* __restrict__ tb,      // [M, 25]
    const float* __restrict__ dbias,   // [512, 25]
    float* __restrict__ outF, int M)
{
    extern __shared__ __bf16 smem[];
    __bf16* Al = smem;                 // 64 x AST2
    __bf16* Bl = smem + BM * AST2;     // 128 x BSTR

    const int tid  = threadIdx.x;
    const int lane = tid & 31;
    const int wave = tid >> 5;
    const int wm = (wave & 1) * 32;
    const int wn = (wave >> 1) * 32;
    const int bm = blockIdx.x * BM;
    const int bn = blockIdx.y * BN;
    const int lrow  = lane & 15;
    const int khalf = (lane >> 4) * 16;

    {   // stage full 64x512 A panel once (256B per thread)
        int r = tid >> 2;
        int c0 = (tid & 3) * 128;
        #pragma unroll
        for (int j = 0; j < 16; ++j)
            cp16(A + (size_t)(bm + r) * HD + c0 + j * 8, Al + r * AST2 + c0 + j * 8);
        cp_wait();
    }
    __syncthreads();

    v8f acc[2][2] = {};

    for (int d = 0; d < 25; ++d) {
        const __bf16* Bd = Wt + (size_t)d * HD * HD;
        v8f cd[2][2] = {};
        for (int k0 = 0; k0 < HD; k0 += BK) {
            {
                int n = tid >> 1;
                int kseg = (tid & 1) * 16;
                const __bf16* src = Bd + (size_t)(bn + n) * HD + k0 + kseg;
                cp16(src,     Bl + n * BSTR + kseg);
                cp16(src + 8, Bl + n * BSTR + kseg + 8);
                cp_wait();
            }
            __syncthreads();
            #pragma unroll
            for (int mi = 0; mi < 2; ++mi) {
                v16bf a = ld_frag16(Al + (wm + mi * 16 + lrow) * AST2 + k0 + khalf);
                #pragma unroll
                for (int ni = 0; ni < 2; ++ni) {
                    v16bf b = ld_frag16(Bl + (wn + ni * 16 + lrow) * BSTR + khalf);
                    cd[mi][ni] = __builtin_amdgcn_wmma_f32_16x16x32_bf16(
                        false, a, false, b, (short)0, cd[mi][ni], false, false);
                }
            }
            __syncthreads();
        }
        // per-sample scale + dynamic bias, accumulated over d
        float bd0 = dbias[(size_t)(bn + wn + lrow) * 25 + d];
        float bd1 = dbias[(size_t)(bn + wn + 16 + lrow) * 25 + d];
        #pragma unroll
        for (int mi = 0; mi < 2; ++mi) {
            #pragma unroll
            for (int v = 0; v < 8; ++v) {
                int row = bm + wm + mi * 16 + v + ((lane >> 4) << 3);
                float s = tb[(size_t)row * 25 + d];
                acc[mi][0][v] += (cd[mi][0][v] + bd0) * s;
                acc[mi][1][v] += (cd[mi][1][v] + bd1) * s;
            }
        }
    }

    #pragma unroll
    for (int mi = 0; mi < 2; ++mi)
        #pragma unroll
        for (int ni = 0; ni < 2; ++ni) {
            int col = bn + wn + ni * 16 + lrow;
            #pragma unroll
            for (int v = 0; v < 8; ++v) {
                int row = bm + wm + mi * 16 + v + ((lane >> 4) << 3);
                outF[(size_t)row * HD + col] = fmaxf(acc[mi][ni][v], 0.f);
            }
        }
}

// ------- dfc2: Q[b] = sum_i h1[b,i]*(sum_d w2[i,d]*tb[b,d]) + sum_d b2[d]*tb[b,d] -------
__global__ __launch_bounds__(64) void dfc2_kernel(
    const float* __restrict__ h1, const float* __restrict__ w2,
    const float* __restrict__ b2, const float* __restrict__ tb,
    float* __restrict__ Q)
{
    __shared__ float red[64];
    __shared__ float tbr[25];
    int b = blockIdx.x;
    int tid = threadIdx.x;
    if (tid < 25) tbr[tid] = tb[(size_t)b * 25 + tid];
    __syncthreads();
    float s = 0.f;
    for (int i = tid; i < HD; i += 64) {
        float w = 0.f;
        #pragma unroll
        for (int d = 0; d < 25; ++d) w += w2[(size_t)i * 25 + d] * tbr[d];
        s += h1[(size_t)b * HD + i] * w;
    }
    red[tid] = s;
    __syncthreads();
    for (int o = 32; o > 0; o >>= 1) {
        if (tid < o) red[tid] += red[tid + o];
        __syncthreads();
    }
    if (tid == 0) {
        float bb = 0.f;
        #pragma unroll
        for (int d = 0; d < 25; ++d) bb += b2[d] * tbr[d];
        Q[b] = red[0] + bb;
    }
}

extern "C" void kernel_launch(void* const* d_in, const int* in_sizes, int n_in,
                              void* d_out, int out_size, void* d_ws, size_t ws_size,
                              hipStream_t stream) {
    const float* t     = (const float*)d_in[0];
    const float* x     = (const float*)d_in[1];
    const float* W1    = (const float*)d_in[2];
    const float* b1    = (const float*)d_in[3];
    const float* W2    = (const float*)d_in[4];
    const float* b2    = (const float*)d_in[5];
    const float* dfc1w = (const float*)d_in[6];
    const float* dfc1b = (const float*)d_in[7];
    const float* dfc2w = (const float*)d_in[8];
    const float* dfc2b = (const float*)d_in[9];

    float* Q       = (float*)d_out;        // [B]
    float* hiddenF = (float*)d_out + NB;   // [B, 512]

    char* ws = (char*)d_ws;
    size_t off = 0;
    auto alloc = [&](size_t bytes) -> void* {
        void* p = ws + off;
        off = (off + bytes + 255) & ~(size_t)255;
        return p;
    };
    float*  tb    = (float*) alloc((size_t)NB * 25 * 4);
    __bf16* xb    = (__bf16*)alloc((size_t)NB * 256 * 2);
    __bf16* W1bT  = (__bf16*)alloc((size_t)HD * 256 * 2);   // [512,256]
    __bf16* W2bT  = (__bf16*)alloc((size_t)HD * HD * 2);    // [512,512]
    __bf16* h1b   = (__bf16*)alloc((size_t)NB * HD * 2);    // relu(x@W1+b1), bf16
    __bf16* hidb  = (__bf16*)alloc((size_t)NB * HD * 2);    // hidden, bf16
    __bf16* wtb   = (__bf16*)alloc((size_t)25 * HD * HD * 2);
    float*  hdfc  = (float*) alloc((size_t)NB * HD * 4);    // h1 of dynamic fc, f32
    (void)ws_size; (void)in_sizes; (void)n_in; (void)out_size;

    basis_kernel<<<NB / 256, 256, 0, stream>>>(t, tb, NB);
    cvt_kernel<<<(NB * 256) / 256, 256, 0, stream>>>(x, xb, (size_t)NB * 256);
    cvt_transpose<<<(HD * 256) / 256, 256, 0, stream>>>(W1, W1bT, 256, HD);
    cvt_transpose<<<(HD * HD) / 256, 256, 0, stream>>>(W2, W2bT, HD, HD);
    cvt_dfc1w<<<(HD * HD) / 256, 256, 0, stream>>>(dfc1w, wtb);

    dim3 grid(NB / BM, HD / BN);
    gemm_bf16<<<grid, 256, 0, stream>>>(xb, W1bT, b1, nullptr, h1b, NB, HD, 256, 1);
    gemm_bf16<<<grid, 256, 0, stream>>>(h1b, W2bT, b2, hiddenF, hidb, NB, HD, HD, 1);

    size_t dfc1_lds = (size_t)(BM * AST2 + BN * BSTR) * sizeof(__bf16);  // 76.8 KB
    dfc1_gemm<<<grid, 256, dfc1_lds, stream>>>(hidb, wtb, tb, dfc1b, hdfc, NB);
    dfc2_kernel<<<NB, 64, 0, stream>>>(hdfc, dfc2w, dfc2b, tb, Q);
}
